// ChebNet_64991445123392
// MI455X (gfx1250) — matrix operand order repaired and checked by reading, compiled, not run on hardware
//
#include <hip/hip_runtime.h>
#include <math.h>

// ---------------------------------------------------------------------------
// ChebNet forward for MI455X (gfx1250, wave32).
//  - Dense cheb GEMMs via v_wmma_f32_16x16x32_bf16, all operands pre-packed
//    bf16 (convert-once); fp32 accumulation.
//  - Sparse L_hat props: fp32 scatter atomics (L2-resident), bf16 gathers.
//  Assumes N % 16 == 0 (N = 50000 = 3125*16 here).
// ---------------------------------------------------------------------------

typedef __attribute__((ext_vector_type(16))) __bf16 v16bf;
typedef __attribute__((ext_vector_type(8)))  __bf16 v8bf;
typedef __attribute__((ext_vector_type(8)))  float  v8f;

#define LDT   224   // padded feature stride = 7*32 K-steps, 14*16 N-tiles
#define HDIM  200   // real hidden width
#define NTILE 14    // 224/16 output column tiles

static __device__ __forceinline__ __bf16 f2bf(float f) {
  unsigned u = __float_as_uint(f);
  u += 0x7fffu + ((u >> 16) & 1u);               // round-to-nearest-even
  return __builtin_bit_cast(__bf16, (unsigned short)(u >> 16));
}

static __device__ __forceinline__ float ldval(const float* p, size_t i) { return p[i]; }
static __device__ __forceinline__ float ldval(const __bf16* p, size_t i) {
  unsigned short s = __builtin_bit_cast(unsigned short, p[i]);
  return __uint_as_float((unsigned)s << 16);
}

// ----------------------------- utility kernels -----------------------------

__global__ void k_zero(float* __restrict__ p, int n) {
  int i = blockIdx.x * blockDim.x + threadIdx.x;
  if (i < n) p[i] = 0.0f;
}

__global__ void k_deg(const long long* __restrict__ src, float* __restrict__ deg, int E) {
  int e = blockIdx.x * blockDim.x + threadIdx.x;
  if (e < E) atomicAdd(&deg[src[e]], 1.0f);
}

__global__ void k_disdiag(const float* __restrict__ deg, const long long* __restrict__ batch,
                          const float* __restrict__ lmax, float* __restrict__ dis,
                          float* __restrict__ diag, int N) {
  int n = blockIdx.x * blockDim.x + threadIdx.x;
  if (n >= N) return;
  float d = deg[n];
  dis[n]  = d > 0.0f ? rsqrtf(fmaxf(d, 1.0f)) : 0.0f;
  diag[n] = 2.0f / lmax[batch[n]] - 1.0f;
}

__global__ void k_w(const long long* __restrict__ src, const long long* __restrict__ dst,
                    const long long* __restrict__ batch, const float* __restrict__ lmax,
                    const float* __restrict__ dis, float* __restrict__ w, int E) {
  int e = blockIdx.x * blockDim.x + threadIdx.x;
  if (e >= E) return;
  long long s = src[e], d = dst[e];
  w[e] = -dis[s] * dis[d] * (2.0f / lmax[batch[s]]);
}

// pack fp32 (row stride LDT) -> compact bf16 (row stride Fk)
__global__ void k_pack(__bf16* __restrict__ dst, const float* __restrict__ src,
                       int Fk, long long total) {
  long long i = (long long)blockIdx.x * blockDim.x + threadIdx.x;
  if (i >= total) return;
  long long n = i / Fk, f = i % Fk;
  dst[i] = f2bf(src[n * LDT + f]);
}

__global__ void k_pack_flat(__bf16* __restrict__ dst, const float* __restrict__ src,
                            long long total) {
  long long i = (long long)blockIdx.x * blockDim.x + threadIdx.x;
  if (i < total) dst[i] = f2bf(src[i]);
}

// t[n,f] = (f<F) ? alpha*diag[n]*hs[n,f] + beta*hb[n,f] : 0     (blockDim = Fk)
template <typename TS, typename TB>
__global__ void cheb_prop_init(float* __restrict__ t,
                               const TS* __restrict__ hs, int ldhs,
                               const TB* __restrict__ hb, int ldhb,
                               const float* __restrict__ diag,
                               float alpha, float beta, int F) {
  const int n = blockIdx.x;
  const int f = threadIdx.x;
  float v = 0.0f;
  if (f < F)
    v = alpha * diag[n] * ldval(hs, (size_t)n * ldhs + f) +
        beta * ldval(hb, (size_t)n * ldhb + f);
  t[(size_t)n * LDT + f] = v;
}

// t[dst[e],f] += coef * w[e] * h[src[e],f]     (one block per edge)
template <typename TH>
__global__ void cheb_scatter(float* __restrict__ t,
                             const TH* __restrict__ h, int ldh,
                             const float* __restrict__ w,
                             const long long* __restrict__ src,
                             const long long* __restrict__ dst,
                             float coef, int F) {
  const int e = blockIdx.x;
  const int f = threadIdx.x;
  if (f >= F) return;
  const long long s = src[e], d = dst[e];
  const float we = coef * w[e];
  atomicAdd(&t[(size_t)d * LDT + f], we * ldval(h, (size_t)s * ldh + f));
}

// Stack + convert weights: Wb[(j*Fk+f), hh] = bf16(W[j,f,hh]) with zero padding.
__global__ void cheb_wb(const float* __restrict__ W, __bf16* __restrict__ Wb,
                        int F, int Fk) {
  int idx = blockIdx.x * blockDim.x + threadIdx.x;
  int tot = 3 * Fk * LDT;
  if (idx >= tot) return;
  int row = idx / LDT, col = idx % LDT;
  int j = row / Fk, f = row % Fk;
  float v = (f < F && col < HDIM) ? W[((size_t)j * F + f) * HDIM + col] : 0.0f;
  Wb[idx] = f2bf(v);
}

// --------------------------- WMMA cheb GEMM --------------------------------
// Cbf[N,224] = bf16( relu( [A0|A1|A2] @ Wb + bias ) ), cols >= 200 forced 0.
// All A operands pre-packed bf16 with common row stride lda (= Fk).
// One wave per 16x16 C tile; K-loop steps of 32 over the three A matrices.
__global__ void cheb_gemm_wmma(const __bf16* __restrict__ A0,
                               const __bf16* __restrict__ A1,
                               const __bf16* __restrict__ A2,
                               int lda,
                               const __bf16* __restrict__ Wb,
                               const float* __restrict__ bias,
                               __bf16* __restrict__ Cbf,
                               int Fk, int Mtiles) {
  const int lane = threadIdx.x & 31;
  const int wave = threadIdx.x >> 5;
  const int tile = blockIdx.x * (blockDim.x >> 5) + wave;
  if (tile >= Mtiles * NTILE) return;               // wave-uniform: EXEC stays all-1s
  const int bm = tile / NTILE;
  const int bn = tile % NTILE;

  const int mrow = bm * 16 + (lane & 15);
  const int hi8  = (lane >> 4) << 3;                // 0 (lanes 0-15) / 8 (lanes 16-31)

  v8f acc;
#pragma unroll
  for (int r = 0; r < 8; ++r) acc[r] = 0.0f;

#pragma unroll
  for (int j = 0; j < 3; ++j) {
    const __bf16* A = (j == 0) ? A0 : ((j == 1) ? A1 : A2);
    const __bf16* arow = A + (size_t)mrow * lda + hi8;
    const __bf16* brow = Wb + (size_t)(j * Fk + hi8) * LDT + bn * 16 + (lane & 15);
    for (int k0 = 0; k0 < Fk; k0 += 32) {
      // A fragment: lane row mrow, K-octets {k0+hi8..+7} and {k0+hi8+16..+23}
      v8bf a_lo = *(const v8bf*)(arow + k0);         // 16B load
      v8bf a_hi = *(const v8bf*)(arow + k0 + 16);    // 16B load
      __builtin_prefetch(arow + k0 + 32, 0, 1);      // global_prefetch_b8
      v16bf a = __builtin_shufflevector(a_lo, a_hi,
                                        0, 1, 2, 3, 4, 5, 6, 7,
                                        8, 9, 10, 11, 12, 13, 14, 15);
      // B fragment: lane holds col N = bn*16+(lane&15), K-octets {k0+hi8..+7} and {k0+hi8+16..+23}
      v16bf b;
#pragma unroll
      for (int s = 0; s < 8; ++s) {
        b[s]     = brow[(size_t)(k0 + s) * LDT];
        b[s + 8] = brow[(size_t)(k0 + 16 + s) * LDT];
      }
      acc = __builtin_amdgcn_wmma_f32_16x16x32_bf16(false, a, false, b,
                                                    (short)0, acc, false, false);
    }
  }

  // C layout: VGPR r -> row bm*16 + r + hi8 ; col = bn*16 + (lane&15)
  const int col   = bn * 16 + (lane & 15);
  const int rbase = bm * 16 + hi8;
  const float bv  = (col < HDIM) ? bias[col] : 0.0f;
#pragma unroll
  for (int r = 0; r < 8; ++r) {
    float v = acc[r] + bv;
    v = v > 0.0f ? v : 0.0f;
    if (col >= HDIM) v = 0.0f;                      // keep K-pad columns zeroed
    Cbf[(size_t)(rbase + r) * LDT + col] = f2bf(v);
  }
}

// ------------------------------ pooling/head -------------------------------

__global__ void k_cnt(const long long* __restrict__ batch, float* __restrict__ cnt, int N) {
  int n = blockIdx.x * blockDim.x + threadIdx.x;
  if (n < N) atomicAdd(&cnt[batch[n]], 1.0f);
}

__global__ void k_pool(const __bf16* __restrict__ h, const long long* __restrict__ batch,
                       float* __restrict__ pooled, int N) {
  int idx = blockIdx.x * blockDim.x + threadIdx.x;
  if (idx >= N * HDIM) return;
  int n = idx / HDIM, f = idx % HDIM;
  float v = ldval(h, (size_t)n * LDT + f);
  int g = (int)batch[n];
  atomicAdd(&pooled[(size_t)g * 2 * HDIM + f], v);
  // post-ReLU values are >= 0, so uint bit order == float order
  atomicMax((unsigned*)&pooled[(size_t)g * 2 * HDIM + HDIM + f], __float_as_uint(v));
}

__global__ void k_head(const float* __restrict__ pooled, const float* __restrict__ cnt,
                       const float* __restrict__ fcw, const float* __restrict__ fcb,
                       float* __restrict__ out, int G) {
  int g = blockIdx.x * blockDim.x + threadIdx.x;
  if (g >= G) return;
  float c = fmaxf(cnt[g], 1.0f);
  float z0 = fcb[0], z1 = fcb[1];
  const float* pg = pooled + (size_t)g * 2 * HDIM;
  for (int k = 0; k < 2 * HDIM; ++k) {
    float v = (k < HDIM) ? pg[k] / c : pg[k];
    z0 += v * fcw[2 * k];
    z1 += v * fcw[2 * k + 1];
  }
  float m = fmaxf(z0, z1);
  float lse = m + logf(expf(z0 - m) + expf(z1 - m));
  out[2 * g]     = z0 - lse;
  out[2 * g + 1] = z1 - lse;
}

// ------------------------------- launcher ----------------------------------

extern "C" void kernel_launch(void* const* d_in, const int* in_sizes, int n_in,
                              void* d_out, int out_size, void* d_ws, size_t ws_size,
                              hipStream_t stream) {
  const float*     x     = (const float*)d_in[0];
  const long long* ei    = (const long long*)d_in[1];
  const long long* batch = (const long long*)d_in[2];
  const float*     lmax  = (const float*)d_in[3];
  const float* Wl[4] = {(const float*)d_in[4], (const float*)d_in[6],
                        (const float*)d_in[8], (const float*)d_in[10]};
  const float* bl[4] = {(const float*)d_in[5], (const float*)d_in[7],
                        (const float*)d_in[9], (const float*)d_in[11]};
  const float* fcw = (const float*)d_in[12];
  const float* fcb = (const float*)d_in[13];

  const int E = in_sizes[1] / 2;
  const int N = in_sizes[2];
  const int G = in_sizes[3];
  const long long* src = ei;
  const long long* dst = ei + E;

  // ---- workspace carve ----
  char* p = (char*)d_ws;
  auto carve = [&](size_t bytes) -> char* {
    char* r = p; p += (bytes + 255) & ~(size_t)255; return r;
  };
  float*  t1     = (float*)carve((size_t)N * LDT * 4);   // fp32 atomic targets
  float*  t2     = (float*)carve((size_t)N * LDT * 4);
  __bf16* t1bf   = (__bf16*)carve((size_t)N * LDT * 2);  // packed GEMM operands
  __bf16* t2bf   = (__bf16*)carve((size_t)N * LDT * 2);
  __bf16* hbfA   = (__bf16*)carve((size_t)N * LDT * 2);  // ping-pong hidden (bf16)
  __bf16* hbfB   = (__bf16*)carve((size_t)N * LDT * 2);
  __bf16* xbf    = (__bf16*)carve((size_t)N * 64 * 2);
  float*  deg    = (float*)carve((size_t)N * 4);
  float*  dis    = (float*)carve((size_t)N * 4);
  float*  diag   = (float*)carve((size_t)N * 4);
  float*  wE     = (float*)carve((size_t)E * 4);
  __bf16* Wb     = (__bf16*)carve((size_t)3 * LDT * LDT * 2);
  float*  pooled = (float*)carve((size_t)G * 2 * HDIM * 4);
  float*  cnt    = (float*)carve((size_t)G * 4);

  // ---- graph normalization + input pack (once) ----
  k_zero<<<(N + 255) / 256, 256, 0, stream>>>(deg, N);
  k_deg<<<(E + 255) / 256, 256, 0, stream>>>(src, deg, E);
  k_disdiag<<<(N + 255) / 256, 256, 0, stream>>>(deg, batch, lmax, dis, diag, N);
  k_w<<<(E + 255) / 256, 256, 0, stream>>>(src, dst, batch, lmax, dis, wE, E);
  {
    long long tot = (long long)N * 64;
    k_pack_flat<<<(unsigned)((tot + 255) / 256), 256, 0, stream>>>(xbf, x, tot);
  }

  // ---- 4 ChebConv(K=3) + ReLU layers ----
  const int Mtiles = (N + 15) / 16;
  const __bf16* hin = xbf;
  int F = 64, Fk = 64;                             // layer 1: K = 64 exact
  __bf16* houts[4] = {hbfA, hbfB, hbfA, hbfB};
  for (int l = 0; l < 4; ++l) {
    const int ldh = (l == 0) ? 64 : LDT;           // hin row stride (bf16)
    // Tx1 = L_hat @ h
    cheb_prop_init<__bf16, __bf16><<<N, Fk, 0, stream>>>(
        t1, hin, ldh, hin, ldh, diag, 1.0f, 0.0f, F);
    cheb_scatter<__bf16><<<E, Fk, 0, stream>>>(t1, hin, ldh, wE, src, dst, 1.0f, F);
    // Tx2 = 2*L_hat @ Tx1 - Tx0
    cheb_prop_init<float, __bf16><<<N, Fk, 0, stream>>>(
        t2, t1, LDT, hin, ldh, diag, 2.0f, -1.0f, F);
    cheb_scatter<float><<<E, Fk, 0, stream>>>(t2, t1, LDT, wE, src, dst, 2.0f, F);
    // pack t1/t2 to compact bf16 (row stride Fk)
    {
      long long tot = (long long)N * Fk;
      unsigned blks = (unsigned)((tot + 255) / 256);
      k_pack<<<blks, 256, 0, stream>>>(t1bf, t1, Fk, tot);
      k_pack<<<blks, 256, 0, stream>>>(t2bf, t2, Fk, tot);
    }
    // stacked bf16 weights
    int wb_tot = 3 * Fk * LDT;
    cheb_wb<<<(wb_tot + 255) / 256, 256, 0, stream>>>(Wl[l], Wb, F, Fk);
    // fused triple GEMM + bias + ReLU via WMMA (bf16 in, bf16 out)
    __bf16* hout = houts[l];
    int blocks = (Mtiles * NTILE + 7) / 8;         // 8 waves / 256-thread block
    cheb_gemm_wmma<<<blocks, 256, 0, stream>>>(hin, t1bf, t2bf, Fk, Wb, bl[l],
                                               hout, Fk, Mtiles);
    hin = hout; F = HDIM; Fk = LDT;                // layers 2-4: K padded 200->224
  }

  // ---- mean+max pool and classifier head ----
  k_zero<<<(G * 2 * HDIM + 255) / 256, 256, 0, stream>>>(pooled, G * 2 * HDIM);
  k_zero<<<(G + 255) / 256, 256, 0, stream>>>(cnt, G);
  k_cnt<<<(N + 255) / 256, 256, 0, stream>>>(batch, cnt, N);
  k_pool<<<(N * HDIM + 255) / 256, 256, 0, stream>>>(hin, batch, pooled, N);
  k_head<<<(G + 255) / 256, 256, 0, stream>>>(pooled, cnt, fcw, fcb, (float*)d_out, G);
}
